// Qwen3MoeSparseMoeBlock_79671643341392
// MI455X (gfx1250) — matrix operand order repaired and checked by reading, not compiled, not run on hardware
//
#include <hip/hip_runtime.h>
#include <hip/hip_bf16.h>
#include <math.h>

#define HDIM 2048
#define IDIM 768
#define NEXP 64
#define TOPK 8
#define NTOK 512

typedef __attribute__((ext_vector_type(16))) __bf16          v16bf;
typedef __attribute__((ext_vector_type(8)))  float           v8f;

// ---- helpers -------------------------------------------------------------

// B-operand fragment: lane holds 8 contiguous K at p, plus 8 contiguous K at p+16
__device__ __forceinline__ v16bf loadB_cvt(const float* __restrict__ p) {
    v16bf r;
#pragma unroll
    for (int i = 0; i < 16; i += 4) {
        float4 q = *(const float4*)(p + (i < 8 ? i : i + 8));
        r[i+0] = (__bf16)q.x; r[i+1] = (__bf16)q.y;
        r[i+2] = (__bf16)q.z; r[i+3] = (__bf16)q.w;
    }
    return r;
}

// A-operand fragment: lane holds 8 contiguous K at p, plus 8 contiguous K at p+16
__device__ __forceinline__ v16bf loadA_cvt(const float* __restrict__ p) {
    v16bf r;
#pragma unroll
    for (int i = 0; i < 8; i += 4) {
        float4 q = *(const float4*)(p + i);
        r[i+0] = (__bf16)q.x; r[i+1] = (__bf16)q.y;
        r[i+2] = (__bf16)q.z; r[i+3] = (__bf16)q.w;
    }
#pragma unroll
    for (int i = 0; i < 8; i += 4) {
        float4 q = *(const float4*)(p + 16 + i);
        r[8+i+0] = (__bf16)q.x; r[8+i+1] = (__bf16)q.y;
        r[8+i+2] = (__bf16)q.z; r[8+i+3] = (__bf16)q.w;
    }
    return r;
}

__device__ __forceinline__ v8f wmma_bf16(v16bf a, v16bf b, v8f c) {
    return __builtin_amdgcn_wmma_f32_16x16x32_bf16(false, a, false, b, (short)0, c, false, false);
}

__device__ __forceinline__ float fast_silu(float g) {
    // g * sigmoid(g) with hw rcp (v_rcp_f32) instead of IEEE divide
    return g * __builtin_amdgcn_rcpf(1.f + __expf(-g));
}

// ---- 1) router: logits -> top-8 of logits -> softmax over the 8 ----------

__global__ __launch_bounds__(64)
void router_kernel(const float* __restrict__ hs, const float* __restrict__ gw,
                   int* __restrict__ topk_idx, float* __restrict__ topk_w) {
    __shared__ float logits[NEXP];
    const int t = blockIdx.x;
    const int e = threadIdx.x;
    const float* x = hs + (size_t)t * HDIM;
    const float* w = gw + (size_t)e * HDIM;
    float acc = 0.f;
    for (int h = 0; h < HDIM; h += 4) {
        float4 xv = *(const float4*)(x + h);
        float4 wv = *(const float4*)(w + h);
        acc += xv.x * wv.x + xv.y * wv.y + xv.z * wv.z + xv.w * wv.w;
    }
    logits[e] = acc;
    __syncthreads();
    if (e == 0) {
        int   idx[TOPK]; float val[TOPK];
        for (int k = 0; k < TOPK; k++) {
            int bi = 0; float bv = -1e30f;
            for (int j = 0; j < NEXP; j++)
                if (logits[j] > bv) { bv = logits[j]; bi = j; }
            idx[k] = bi; val[k] = bv; logits[bi] = -1e30f;
        }
        float m = val[0], s = 0.f, wt[TOPK];
        for (int k = 0; k < TOPK; k++) { wt[k] = __expf(val[k] - m); s += wt[k]; }
        float inv = 1.f / s;
        for (int k = 0; k < TOPK; k++) {
            topk_idx[t * TOPK + k] = idx[k];
            topk_w  [t * TOPK + k] = wt[k] * inv;
        }
    }
}

// ---- 2) deterministic per-expert pair lists ------------------------------

__global__ void build_lists_kernel(const int* __restrict__ topk_idx,
                                   int* __restrict__ counts, int* __restrict__ lists) {
    const int e = blockIdx.x;
    if (threadIdx.x != 0) return;
    int c = 0;
    for (int p = 0; p < NTOK * TOPK; p++)
        if (topk_idx[p] == e) lists[e * NTOK + c++] = p;   // p = t*8 + k, token order
    counts[e] = c;
}

// ---- 3) GEMM1: gathered X @ gate_up^T, fused SiLU*up ---------------------
// grid = (IDIM/128, NTOK/64, NEXP), block = 256 (8 waves)
// wave computes 64 rows x 16 gate-cols (and the paired up-cols)

__global__ __launch_bounds__(256)
void gemm1_kernel(const float* __restrict__ hs, const float* __restrict__ guw,
                  const int* __restrict__ counts, const int* __restrict__ lists,
                  float* __restrict__ act) {
    const int e   = blockIdx.z;
    const int cnt = counts[e];
    const int m0  = blockIdx.y * 64;
    if (m0 >= cnt) return;

    const int lane = threadIdx.x & 31;
    const int wave = threadIdx.x >> 5;
    const int ncol  = (blockIdx.x * 8 + wave) * 16 + (lane & 15);
    const int koffB = (lane >> 4) * 8;
    const int koffA = (lane >> 4) * 8;

    const float* W  = guw + (size_t)e * (2 * IDIM) * HDIM;
    const float* Bg = W + (size_t)ncol * HDIM + koffB;
    const float* Bu = W + (size_t)(ncol + IDIM) * HDIM + koffB;

    const float* Ap[4];
#pragma unroll
    for (int mt = 0; mt < 4; mt++) {
        int r  = m0 + mt * 16 + (lane & 15);
        int rc = r < cnt ? r : cnt - 1;
        int pr = lists[e * NTOK + rc];
        Ap[mt] = hs + (size_t)(pr >> 3) * HDIM + koffA;
    }

    v8f cg[4] = {}, cu[4] = {};
    for (int kk = 0; kk < HDIM; kk += 32) {
        __builtin_prefetch(Bg + kk + 256, 0, 0);
        __builtin_prefetch(Bu + kk + 256, 0, 0);
        v16bf bg = loadB_cvt(Bg + kk);
        v16bf bu = loadB_cvt(Bu + kk);
#pragma unroll
        for (int mt = 0; mt < 4; mt++) {
            v16bf a = loadA_cvt(Ap[mt] + kk);
            cg[mt] = wmma_bf16(a, bg, cg[mt]);
            cu[mt] = wmma_bf16(a, bu, cu[mt]);
        }
    }

    // epilogue: act = silu(gate) * up, scattered to pair rows
#pragma unroll
    for (int mt = 0; mt < 4; mt++) {
#pragma unroll
        for (int i = 0; i < 8; i++) {
            int r = m0 + mt * 16 + i + (lane >> 4) * 8;
            if (r < cnt) {
                int pr = lists[e * NTOK + r];
                act[(size_t)pr * IDIM + ncol] = fast_silu(cg[mt][i]) * cu[mt][i];
            }
        }
    }
}

// ---- 4) GEMM2: act @ down_w^T -------------------------------------------
// grid = (HDIM/128, NTOK/64, NEXP), block = 256

__global__ __launch_bounds__(256)
void gemm2_kernel(const float* __restrict__ act, const float* __restrict__ dw,
                  const int* __restrict__ counts, const int* __restrict__ lists,
                  float* __restrict__ dout) {
    const int e   = blockIdx.z;
    const int cnt = counts[e];
    const int m0  = blockIdx.y * 64;
    if (m0 >= cnt) return;

    const int lane = threadIdx.x & 31;
    const int wave = threadIdx.x >> 5;
    const int hcol = (blockIdx.x * 8 + wave) * 16 + (lane & 15);
    const int koffB = (lane >> 4) * 8;
    const int koffA = (lane >> 4) * 8;

    const float* Bp = dw + (size_t)e * HDIM * IDIM + (size_t)hcol * IDIM + koffB;

    const float* Ap[4];
#pragma unroll
    for (int mt = 0; mt < 4; mt++) {
        int r  = m0 + mt * 16 + (lane & 15);
        int rc = r < cnt ? r : cnt - 1;
        int pr = lists[e * NTOK + rc];
        Ap[mt] = act + (size_t)pr * IDIM + koffA;
    }

    v8f c[4] = {};
    for (int kk = 0; kk < IDIM; kk += 32) {
        __builtin_prefetch(Bp + kk + 256, 0, 0);
        v16bf b = loadB_cvt(Bp + kk);
#pragma unroll
        for (int mt = 0; mt < 4; mt++) {
            v16bf a = loadA_cvt(Ap[mt] + kk);
            c[mt] = wmma_bf16(a, b, c[mt]);
        }
    }

#pragma unroll
    for (int mt = 0; mt < 4; mt++) {
#pragma unroll
        for (int i = 0; i < 8; i++) {
            int r = m0 + mt * 16 + i + (lane >> 4) * 8;
            if (r < cnt) {
                int pr = lists[e * NTOK + r];
                dout[(size_t)pr * HDIM + hcol] = c[mt][i];
            }
        }
    }
}

// ---- 5) weighted combine (fixed order => deterministic) ------------------

__global__ __launch_bounds__(256)
void combine_kernel(const float* __restrict__ dout, const float* __restrict__ topk_w,
                    float* __restrict__ out) {
    int idx = blockIdx.x * blockDim.x + threadIdx.x;   // T*H threads
    int t = idx >> 11;          // / HDIM
    int h = idx & (HDIM - 1);
    float s = 0.f;
#pragma unroll
    for (int k = 0; k < TOPK; k++)
        s += topk_w[t * TOPK + k] * dout[(size_t)(t * TOPK + k) * HDIM + h];
    out[idx] = s;
}

// ---- launch --------------------------------------------------------------

extern "C" void kernel_launch(void* const* d_in, const int* in_sizes, int n_in,
                              void* d_out, int out_size, void* d_ws, size_t ws_size,
                              hipStream_t stream) {
    const float* hs     = (const float*)d_in[0];   // (T, H)
    const float* gate_w = (const float*)d_in[1];   // (E, H)
    const float* guw    = (const float*)d_in[2];   // (E, 2I, H)
    const float* dw     = (const float*)d_in[3];   // (E, H, I)
    float* out = (float*)d_out;

    char* ws = (char*)d_ws;
    int*   topk_idx = (int*)  (ws);                         // 4096 ints
    float* topk_w   = (float*)(ws + 16384);                 // 4096 floats
    int*   counts   = (int*)  (ws + 32768);                 // 64 ints (padded)
    int*   lists    = (int*)  (ws + 33792);                 // 64*512 ints
    float* act      = (float*)(ws + 164864);                // 4096*768 floats
    float* dout     = (float*)(ws + 12747776ull);           // 4096*2048 floats

    router_kernel<<<NTOK, 64, 0, stream>>>(hs, gate_w, topk_idx, topk_w);
    build_lists_kernel<<<NEXP, 32, 0, stream>>>(topk_idx, counts, lists);
    gemm1_kernel<<<dim3(IDIM / 128, NTOK / 64, NEXP), 256, 0, stream>>>(hs, guw, counts, lists, act);
    gemm2_kernel<<<dim3(HDIM / 128, NTOK / 64, NEXP), 256, 0, stream>>>(act, dw, counts, lists, dout);
    combine_kernel<<<(NTOK * HDIM) / 256, 256, 0, stream>>>(dout, topk_w, out);
}
